// Loss_26886495273741
// MI455X (gfx1250) — matrix operand order repaired and checked by reading, not compiled, not run on hardware
//
#include <hip/hip_runtime.h>
#include <math.h>

typedef _Float16 h16 __attribute__((ext_vector_type(16)));
typedef _Float16 h8  __attribute__((ext_vector_type(8)));
typedef float    f8  __attribute__((ext_vector_type(8)));

// async-to-LDS builtin operand type: int vector of 4 (b128 payload)
typedef int av4i __attribute__((vector_size(16)));
typedef __attribute__((address_space(1))) av4i* g4p;   // global b128 pointer
typedef __attribute__((address_space(3))) av4i* l4p;   // LDS b128 pointer

#define FEAT_D        256
#define TILE_COLS     16
#define WAVES         4
#define BLOCK_THREADS (WAVES * 32)
#define STRIPS        2                                  // row strips per wave
#define ROWS_PER_WAVE (16 * STRIPS)                      // 32
#define ROWS_PER_BLOCK (WAVES * ROWS_PER_WAVE)           // 128
#define COL_SPLIT     2                                  // column-space split
#define KCH           (FEAT_D / 32)                      // 8 K-chunks of 32
// LDS row stride in halves: 264 halves = 528 B; 528 % 256 = 16 -> lane l hits
// bank offset 16*l mod 256: all 64 banks covered exactly twice per wave
// request = conflict-free (512-byte stride was a full 16-way conflict).
#define STR           (FEAT_D + 8)

#if defined(__has_builtin)
# if __has_builtin(__builtin_amdgcn_global_load_async_to_lds_b128)
#  define HAVE_ASYNC_LDS 1
# endif
#endif

__device__ __forceinline__ void wait_async_all() {
#if defined(HAVE_ASYNC_LDS)
# if __has_builtin(__builtin_amdgcn_s_wait_asynccnt)
  __builtin_amdgcn_s_wait_asynccnt(0);
# else
  asm volatile("s_wait_asynccnt 0" ::: "memory");
# endif
#endif
}

// ---------------------------------------------------------------------------
// Kernel 0: split fp32 feature into f16 hi/lo pair (hi + lo ~= x, 22 mantissa
// bits) so the GEMM runs on the f16 matrix pipe at fp32-like ranking accuracy.
// ---------------------------------------------------------------------------
__global__ void split_f16_kernel(const float* __restrict__ x,
                                 _Float16* __restrict__ hi,
                                 _Float16* __restrict__ lo, int total) {
  int i = blockIdx.x * blockDim.x + threadIdx.x;
  if (i < total) {
    float v = x[i];
    _Float16 h = (_Float16)v;
    hi[i] = h;
    lo[i] = (_Float16)(v - (float)h);
  }
}

// ---------------------------------------------------------------------------
// Kernel 1: fused GEMM (F * F^T via 3x f16 WMMA emulation) + row-wise argmax.
// Each wave owns 2 strips of 16 rows (A frags VGPR-resident for all K=256);
// each staged B fragment now feeds 6 WMMAs (48 WMMA : 32 ds_load per tile).
// Grid is (row blocks) x (2 column halves); halves are combined in kernel 2.
// ---------------------------------------------------------------------------
__global__ __launch_bounds__(BLOCK_THREADS, 1)
void nn_argmax_kernel(const _Float16* __restrict__ Fhi,
                      const _Float16* __restrict__ Flo,
                      float* __restrict__ bestV, int* __restrict__ bestI,
                      int n) {
  __shared__ __attribute__((aligned(32))) _Float16 LBH[2][TILE_COLS * STR];
  __shared__ __attribute__((aligned(32))) _Float16 LBL[2][TILE_COLS * STR];

  const int lane = threadIdx.x & 31;
  const int wave = threadIdx.x >> 5;
  const int h_   = lane >> 4;    // 0 for lanes 0-15, 1 for lanes 16-31
  const int l16  = lane & 15;
  const int half = blockIdx.y;   // which half of the column space we scan
  const int r0   = blockIdx.x * ROWS_PER_BLOCK + wave * ROWS_PER_WAVE;

  // ---- A fragments: 2 strips x 16 rows x 256 K, hi+lo, in registers.
  // ISA 7.12.2 (16-bit A 16x32): lane m holds K = k0+8*h_..+7 and
  // K = k0+16+8*h_..+7 as two contiguous 16B chunks.
  h16 a_hi[STRIPS][KCH], a_lo[STRIPS][KCH];
  for (int s = 0; s < STRIPS; ++s) {
    const _Float16* hrow = Fhi + (size_t)(r0 + s * 16 + l16) * FEAT_D;
    const _Float16* lrow = Flo + (size_t)(r0 + s * 16 + l16) * FEAT_D;
    for (int kk = 0; kk < KCH; ++kk) {
      union { h16 v; struct { h8 a, b; } p; } u, w;
      u.p.a = *(const h8*)(hrow + kk * 32 + h_ * 8);
      u.p.b = *(const h8*)(hrow + kk * 32 + 16 + h_ * 8);
      a_hi[s][kk] = u.v;
      w.p.a = *(const h8*)(lrow + kk * 32 + h_ * 8);
      w.p.b = *(const h8*)(lrow + kk * 32 + 16 + h_ * 8);
      a_lo[s][kk] = w.v;
    }
  }

  float best[STRIPS][8];
  int   bidx[STRIPS][8];
  for (int s = 0; s < STRIPS; ++s)
    for (int v = 0; v < 8; ++v) { best[s][v] = -3.402823466e38f; bidx[s][v] = 0; }

  const int colBase0 = half * (n / COL_SPLIT);
  const int nTiles   = (n / COL_SPLIT) / TILE_COLS;

  auto stage = [&](int b, int cb) {
    // 16 dot-columns == 16 feature rows (K contiguous), hi+lo = 16 KB.
#if defined(HAVE_ASYNC_LDS)
    for (int e = threadIdx.x; e < TILE_COLS * FEAT_D / 8; e += BLOCK_THREADS) {
      int row = e >> 5;          // 32 x 16B segments per source row
      int off = (e & 31) * 8;    // offset in halves
      __builtin_amdgcn_global_load_async_to_lds_b128(
          (g4p)(void*)const_cast<_Float16*>(Fhi + (size_t)(cb + row) * FEAT_D + off),
          (l4p)(void*)&LBH[b][row * STR + off], 0, 0);
      __builtin_amdgcn_global_load_async_to_lds_b128(
          (g4p)(void*)const_cast<_Float16*>(Flo + (size_t)(cb + row) * FEAT_D + off),
          (l4p)(void*)&LBL[b][row * STR + off], 0, 0);
    }
#else
    for (int e = threadIdx.x; e < TILE_COLS * FEAT_D / 8; e += BLOCK_THREADS) {
      int row = e >> 5;
      int off = e & 31;
      *(uint4*)&LBH[b][row * STR + off * 8] =
          ((const uint4*)(Fhi + (size_t)(cb + row) * FEAT_D))[off];
      *(uint4*)&LBL[b][row * STR + off * 8] =
          ((const uint4*)(Flo + (size_t)(cb + row) * FEAT_D))[off];
    }
#endif
  };

  stage(0, colBase0);

  for (int t = 0; t < nTiles; ++t) {
    const int buf = t & 1;
    const int cb  = colBase0 + t * TILE_COLS;

    wait_async_all();
    __syncthreads();
    if (t + 1 < nTiles) stage(buf ^ 1, cb + TILE_COLS);  // prefetch overlaps compute

    // ---- two 16x16 C tiles over K=256: per K-chunk, one bh/bl fetch feeds
    // 6 WMMAs (hi*hi + hi*lo + lo*hi for each strip).
    f8 c0 = {}, c1 = {};
    for (int kk = 0; kk < KCH; ++kk) {
      // ISA 7.12.2 (16-bit B 32x16): lane (col=l16) holds K = k0+8*h_..+7 and
      // K = k0+16+8*h_..+7 (same as A). Two explicit 16B pieces.
      union { h16 v; struct { h8 a, b; } p; } ub, wb;
      const _Float16* ph = &LBH[buf][l16 * STR + kk * 32 + h_ * 8];
      const _Float16* pl = &LBL[buf][l16 * STR + kk * 32 + h_ * 8];
      ub.p.a = *(const h8*)(ph);
      ub.p.b = *(const h8*)(ph + 16);
      wb.p.a = *(const h8*)(pl);
      wb.p.b = *(const h8*)(pl + 16);
      const h16 bh = ub.v, bl = wb.v;
      c0 = __builtin_amdgcn_wmma_f32_16x16x32_f16(false, a_hi[0][kk], false, bh,
                                                  (short)0, c0, false, false);
      c0 = __builtin_amdgcn_wmma_f32_16x16x32_f16(false, a_hi[0][kk], false, bl,
                                                  (short)0, c0, false, false);
      c0 = __builtin_amdgcn_wmma_f32_16x16x32_f16(false, a_lo[0][kk], false, bh,
                                                  (short)0, c0, false, false);
      c1 = __builtin_amdgcn_wmma_f32_16x16x32_f16(false, a_hi[1][kk], false, bh,
                                                  (short)0, c1, false, false);
      c1 = __builtin_amdgcn_wmma_f32_16x16x32_f16(false, a_hi[1][kk], false, bl,
                                                  (short)0, c1, false, false);
      c1 = __builtin_amdgcn_wmma_f32_16x16x32_f16(false, a_lo[1][kk], false, bh,
                                                  (short)0, c1, false, false);
    }

    // ---- fused argmax update. C layout: VGPR v -> row rs+v+8*h_, col cb+l16.
    const int col = cb + l16;
    for (int s = 0; s < STRIPS; ++s) {
      const f8& c = (s == 0) ? c0 : c1;
      const int rs = r0 + s * 16;
      if (cb == rs) {  // wave-uniform: only the tile containing the diagonal
        for (int v = 0; v < 8; ++v) {
          float val = c[v];
          if (col == rs + v + 8 * h_) val = -3.402823466e38f;  // mask self
          if (val > best[s][v]) { best[s][v] = val; bidx[s][v] = col; }
        }
      } else {
        for (int v = 0; v < 8; ++v) {
          float val = c[v];
          if (val > best[s][v]) { best[s][v] = val; bidx[s][v] = col; }
        }
      }
    }
    __syncthreads();  // everyone done reading buf before it is restaged
  }

  // ---- reduce argmax across the 16 lanes sharing each row; prefer smaller
  // index on ties (jnp.argmax semantics).
  for (int s = 0; s < STRIPS; ++s) {
    for (int v = 0; v < 8; ++v) {
      for (int mask = 8; mask >= 1; mask >>= 1) {
        float ov = __shfl_xor(best[s][v], mask, 32);
        int   oi = __shfl_xor(bidx[s][v], mask, 32);
        if (ov > best[s][v] || (ov == best[s][v] && oi < bidx[s][v])) {
          best[s][v] = ov; bidx[s][v] = oi;
        }
      }
    }
    const int rs = r0 + s * 16;
    if (lane == 0)
      for (int v = 0; v < 8; ++v) {
        bestV[half * n + rs + v] = best[s][v];
        bestI[half * n + rs + v] = bidx[s][v];
      }
    if (lane == 16)
      for (int v = 0; v < 8; ++v) {
        bestV[half * n + rs + 8 + v] = best[s][v];
        bestI[half * n + rs + 8 + v] = bidx[s][v];
      }
  }
}

// ---------------------------------------------------------------------------
// Kernel 2: combine column halves + distances + loss. One wave per row.
// Half-0 indices are always lower than half-1, so strict > keeps the first
// maximum. loss = -mean(log(n * ||x_i - x_{I(i)} + eps||_2))
// ---------------------------------------------------------------------------
__global__ void loss_kernel(const float* __restrict__ x,
                            const float* __restrict__ bestV,
                            const int* __restrict__ bestI,
                            float* __restrict__ out, int n) {
  int row  = (blockIdx.x * blockDim.x + threadIdx.x) >> 5;
  int lane = threadIdx.x & 31;
  if (row >= n) return;
  float b0 = bestV[row], b1 = bestV[n + row];
  int   i0 = bestI[row], i1 = bestI[n + row];
  int   nn = (b1 > b0) ? i1 : i0;
  const float* a = x + (size_t)row * FEAT_D;
  const float* b = x + (size_t)nn * FEAT_D;
  float s = 0.f;
  for (int k = lane; k < FEAT_D; k += 32) {
    float d = a[k] - b[k] + 1e-6f;  // eps added elementwise, per reference
    s += d * d;
  }
  for (int mask = 16; mask >= 1; mask >>= 1) s += __shfl_xor(s, mask, 32);
  if (lane == 0) atomicAdd(out, -logf((float)n * sqrtf(s)) / (float)n);
}

// ---------------------------------------------------------------------------
extern "C" void kernel_launch(void* const* d_in, const int* in_sizes, int n_in,
                              void* d_out, int out_size, void* d_ws, size_t ws_size,
                              hipStream_t stream) {
  const float* feat = (const float*)d_in[0];
  const int total = in_sizes[0];        // 16384 * 256
  const int n = total / FEAT_D;         // 16384

  _Float16* hi = (_Float16*)d_ws;
  _Float16* lo = hi + total;
  float* bv = (float*)(lo + total);     // [2][n] per-half best values
  int*   bi = (int*)(bv + 2 * (size_t)n);  // [2][n] per-half best indices

  split_f16_kernel<<<(total + 255) / 256, 256, 0, stream>>>(feat, hi, lo, total);

  dim3 grid(n / ROWS_PER_BLOCK, COL_SPLIT);
  nn_argmax_kernel<<<grid, BLOCK_THREADS, 0, stream>>>(hi, lo, bv, bi, n);

  (void)hipMemsetAsync(d_out, 0, sizeof(float), stream);
  loss_kernel<<<(n * 32 + 255) / 256, 256, 0, stream>>>(feat, bv, bi,
                                                        (float*)d_out, n);
}
